// DMDG_12360915878242
// MI455X (gfx1250) — matrix operand order repaired and checked by reading, not compiled, not run on hardware
//
#include <hip/hip_runtime.h>
#include <hip/hip_bf16.h>
#include <math.h>

// ---------------- model constants ----------------
#define BATCH 8
#define NFC   64
#define HHH   128
#define WWW   128
#define HWN   16384          // H*W
#define HEADS 4

typedef __attribute__((ext_vector_type(16))) _Float16 v16h;
typedef __attribute__((ext_vector_type(8)))  float    v8f;
typedef __attribute__((ext_vector_type(4)))  unsigned int u32x4;
typedef __attribute__((ext_vector_type(8)))  int      i32x8;
typedef __attribute__((ext_vector_type(4)))  int      i32x4;

// =====================================================================
// f32 -> f16 convert
// =====================================================================
__global__ __launch_bounds__(256) void k_cvt(const float* __restrict__ x,
                                             _Float16* __restrict__ y, int n) {
  int i = blockIdx.x * 256 + threadIdx.x;
  if (i < n) y[i] = (_Float16)x[i];
}

// =====================================================================
// pack f32 weight [Cout x Cin] into WMMA 16x16x32 A-fragment layout
//   storage: [mt][kt][lane(32)][half(16)]
//   lane<16 : M=lane,    K = h<8 ? h   : h+8
//   lane>=16: M=lane-16, K = h<8 ? h+8 : h+16      (K += kt*32)
// =====================================================================
__global__ __launch_bounds__(256) void k_pack(const float* __restrict__ w,
                                              _Float16* __restrict__ ap,
                                              int Cout, int Cin, int Kt) {
  int idx = blockIdx.x * 256 + threadIdx.x;
  int total = (Cout / 16) * Kt * 512;
  if (idx >= total) return;
  int h    = idx & 15;
  int lane = (idx >> 4) & 31;
  int frag = idx >> 9;
  int mt = frag / Kt, kt = frag % Kt;
  int M = lane & 15;
  int K = (lane < 16) ? ((h < 8) ? h : h + 8) : ((h < 8) ? h + 8 : h + 16);
  K += kt * 32;
  float v = (K < Cin) ? w[(mt * 16 + M) * Cin + K] : 0.f;
  ap[idx] = (_Float16)v;
}

// =====================================================================
// 1x1 conv as WMMA GEMM.  Y[cout,pix] = sum_cin W * X  (+bias, act)
// One block: one (batch, 128-pixel slab). TDM loads [Cin x 128] f16 tile
// into LDS; 8 waves each own a 16-pixel column slab; MT = Cout/16 tiles.
// =====================================================================
template <int MT>
__global__ __launch_bounds__(256) void k_conv1x1(
    const _Float16* __restrict__ X, int XCb,
    const _Float16* __restrict__ Apack, int Kt, int Cin,
    const float* __restrict__ bias,
    _Float16* __restrict__ Y, int YCb, int act) {
  __shared__ _Float16 tile[256 * 128];     // up to Cin=256
  const int b  = blockIdx.y;
  const int n0 = blockIdx.x * 128;
  const _Float16* xbase = X + (size_t)b * XCb * HWN + n0;

#if __has_builtin(__builtin_amdgcn_tensor_load_to_lds) && __has_builtin(__builtin_amdgcn_s_wait_tensorcnt)
  if (threadIdx.x < 32) {                  // wave 0 issues the TDM DMA
    unsigned long long ga = (unsigned long long)(uintptr_t)xbase;
    unsigned lds_off = (unsigned)(uintptr_t)&tile[0];
    unsigned td0 = HWN;                    // tensor dim0 (row length)
    u32x4 g0;
    g0[0] = 1u;                                              // count=1
    g0[1] = lds_off;                                         // lds_addr
    g0[2] = (unsigned)(ga & 0xFFFFFFFFu);                    // gaddr lo
    g0[3] = (unsigned)((ga >> 32) & 0x1FFFFFFu) | 0x80000000u; // hi | type=2
    i32x8 g1;
    g1[0] = (int)(1u << 16);                                 // data_size=2B
    g1[1] = (int)((td0 & 0xFFFFu) << 16);                    // td0 lo16
    g1[2] = (int)((td0 >> 16) | ((unsigned)(Cin & 0xFFFF) << 16)); // td0 hi | td1 lo
    g1[3] = (int)(((unsigned)Cin >> 16) | (128u << 16));     // td1 hi | tile0=128
    g1[4] = (int)(Cin & 0xFFFF);                             // tile1=Cin, tile2=0
    g1[5] = (int)td0;                                        // dim0_stride lo32
    g1[6] = 0; g1[7] = 0;
    i32x4 z4 = {0, 0, 0, 0};
    i32x8 z8 = {0, 0, 0, 0, 0, 0, 0, 0};
    __builtin_amdgcn_tensor_load_to_lds(g0, g1, z4, z4, z8, 0);
    __builtin_amdgcn_s_wait_tensorcnt(0);
  }
#else
  for (int i = threadIdx.x; i < Cin * 128; i += 256) {
    int r = i >> 7, c = i & 127;
    tile[i] = xbase[(size_t)r * HWN + c];
  }
#endif
  __syncthreads();

  const int wave = threadIdx.x >> 5, lane = threadIdx.x & 31;
  const int ncol = lane & 15;
  const int kbase = (lane >> 4) << 3;      // 0 or 8
  v8f zero = {0.f, 0.f, 0.f, 0.f, 0.f, 0.f, 0.f, 0.f};
  v8f acc[MT];
#pragma unroll
  for (int m = 0; m < MT; ++m) acc[m] = zero;

  for (int kt = 0; kt < Kt; ++kt) {
    const int K0 = kt * 32 + kbase;
    v16h bf;
#pragma unroll
    for (int h = 0; h < 16; ++h) {
      int K = K0 + h + ((h < 8) ? 0 : 8);
      bf[h] = (K < Cin) ? tile[K * 128 + (wave << 4) + ncol] : (_Float16)0.f;
    }
    const _Float16* ap = Apack + (size_t)kt * 512 + lane * 16;
#pragma unroll
    for (int m = 0; m < MT; ++m) {
      v16h af = *(const v16h*)(ap + (size_t)m * Kt * 512);
      acc[m] = __builtin_amdgcn_wmma_f32_16x16x32_f16(
          false, af, false, bf, (short)0, acc[m], false, false);
    }
  }

  const int pix = n0 + (wave << 4) + ncol;
  const int moff = (lane >= 16) ? 8 : 0;
#pragma unroll
  for (int m = 0; m < MT; ++m) {
#pragma unroll
    for (int r = 0; r < 8; ++r) {
      int cout = m * 16 + r + moff;
      float v = acc[m][r] + (bias ? bias[cout] : 0.f);
      if (act) v = (v >= 0.f) ? v : 0.1f * v;
      Y[((size_t)b * YCb + cout) * HWN + pix] = (_Float16)v;
    }
  }
}

// =====================================================================
// depthwise 3x3 (static per-channel or dynamic per-(b,c) kernel)
// =====================================================================
__global__ __launch_bounds__(256) void k_dw3x3(
    const _Float16* __restrict__ X, int XCb, const float* __restrict__ Wk,
    const float* __restrict__ bias, _Float16* __restrict__ Y, int YCb,
    int C, int act, int dyn) {
  int bc = blockIdx.y;
  int b = bc / C, c = bc % C;
  const float* w = Wk + (size_t)(dyn ? bc : c) * 9;
  int p = blockIdx.x * 256 + threadIdx.x;
  int y = p >> 7, x = p & 127;
  const _Float16* src = X + ((size_t)b * XCb + c) * HWN;
  float s = bias ? bias[c] : 0.f;
#pragma unroll
  for (int ky = 0; ky < 3; ++ky) {
    int iy = y + ky - 1;
    if (iy < 0 || iy >= HHH) continue;
#pragma unroll
    for (int kx = 0; kx < 3; ++kx) {
      int ix = x + kx - 1;
      if (ix < 0 || ix >= WWW) continue;
      s += w[ky * 3 + kx] * (float)src[iy * WWW + ix];
    }
  }
  if (act) s = (s >= 0.f) ? s : 0.1f * s;
  Y[((size_t)b * YCb + c) * HWN + p] = (_Float16)s;
}

// =====================================================================
// dense 3x3 conv (only 16->16, lfde.c3)
// =====================================================================
__global__ __launch_bounds__(256) void k_conv3x3(
    const _Float16* __restrict__ X, int XCb, const float* __restrict__ W,
    const float* __restrict__ bias, _Float16* __restrict__ Y, int YCb,
    int Ci, int Co, int act) {
  int bco = blockIdx.y;
  int b = bco / Co, co = bco % Co;
  int p = blockIdx.x * 256 + threadIdx.x;
  int y = p >> 7, x = p & 127;
  float s = bias ? bias[co] : 0.f;
  for (int ci = 0; ci < Ci; ++ci) {
    const _Float16* src = X + ((size_t)b * XCb + ci) * HWN;
    const float* w = W + ((size_t)co * Ci + ci) * 9;
#pragma unroll
    for (int ky = 0; ky < 3; ++ky) {
      int iy = y + ky - 1;
      if (iy < 0 || iy >= HHH) continue;
#pragma unroll
      for (int kx = 0; kx < 3; ++kx) {
        int ix = x + kx - 1;
        if (ix < 0 || ix >= WWW) continue;
        s += w[ky * 3 + kx] * (float)src[iy * WWW + ix];
      }
    }
  }
  if (act) s = (s >= 0.f) ? s : 0.1f * s;
  Y[((size_t)b * YCb + co) * HWN + p] = (_Float16)s;
}

// =====================================================================
// per-(b,c) reduction over HW: mode 0 -> mean, mode 1 -> 1/max(||x||2,eps)
// =====================================================================
__global__ __launch_bounds__(256) void k_gap(const _Float16* __restrict__ X,
                                             int XCb, int C,
                                             float* __restrict__ out, int mode) {
  int bc = blockIdx.x;
  int b = bc / C, c = bc % C;
  const _Float16* x = X + ((size_t)b * XCb + c) * HWN;
  float s = 0.f;
  for (int i = threadIdx.x; i < HWN; i += 256) {
    float v = (float)x[i];
    s += mode ? v * v : v;
  }
  __shared__ float red[256];
  red[threadIdx.x] = s;
  __syncthreads();
  for (int st = 128; st > 0; st >>= 1) {
    if (threadIdx.x < st) red[threadIdx.x] += red[threadIdx.x + st];
    __syncthreads();
  }
  if (threadIdx.x == 0)
    out[bc] = mode ? 1.f / fmaxf(sqrtf(red[0]), 1e-12f) : red[0] / (float)HWN;
}

// small matvec: s[b,co] = bias[co] + sum_ci w[co,ci]*pooled[b,ci]
__global__ __launch_bounds__(256) void k_scavec(const float* __restrict__ pooled,
                                                const float* __restrict__ w,
                                                const float* __restrict__ bias,
                                                float* __restrict__ s,
                                                int Ci, int Co) {
  int t = blockIdx.x * 256 + threadIdx.x;
  if (t >= BATCH * Co) return;
  int b = t / Co, co = t % Co;
  float acc = bias ? bias[co] : 0.f;
  for (int ci = 0; ci < Ci; ++ci) acc += w[co * Ci + ci] * pooled[b * Ci + ci];
  s[t] = acc;
}

__global__ __launch_bounds__(256) void k_scale(_Float16* __restrict__ X, int XCb,
                                               int C, const float* __restrict__ s,
                                               int act) {
  int bc = blockIdx.y;
  int b = bc / C, c = bc % C;
  int p = blockIdx.x * 256 + threadIdx.x;
  size_t o = ((size_t)b * XCb + c) * HWN + p;
  float v = (float)X[o] * s[bc];
  if (act) v = (v >= 0.f) ? v : 0.1f * v;
  X[o] = (_Float16)v;
}

// copy / add between channel views
__global__ __launch_bounds__(256) void k_axpy(_Float16* __restrict__ D, int DCb,
                                              const _Float16* __restrict__ S,
                                              int SCb, int C, int add) {
  int bc = blockIdx.y;
  int b = bc / C, c = bc % C;
  int p = blockIdx.x * 256 + threadIdx.x;
  size_t od = ((size_t)b * DCb + c) * HWN + p;
  size_t os = ((size_t)b * SCb + c) * HWN + p;
  float v = (float)S[os] + (add ? (float)D[od] : 0.f);
  D[od] = (_Float16)v;
}

// bias-free LayerNorm over channels
__global__ __launch_bounds__(256) void k_ln(const _Float16* __restrict__ X,
                                            const float* __restrict__ g,
                                            _Float16* __restrict__ Y, int C) {
  int b = blockIdx.y;
  int p = blockIdx.x * 256 + threadIdx.x;
  const _Float16* x = X + (size_t)b * C * HWN + p;
  float s = 0.f, ss = 0.f;
  for (int c = 0; c < C; ++c) {
    float v = (float)x[(size_t)c * HWN];
    s += v; ss += v * v;
  }
  float mean = s / (float)C;
  float var = ss / (float)C - mean * mean;
  float inv = rsqrtf(var + 1e-5f);
  _Float16* y = Y + (size_t)b * C * HWN + p;
  for (int c = 0; c < C; ++c)
    y[(size_t)c * HWN] = (_Float16)((float)x[(size_t)c * HWN] * inv * g[c]);
}

// =====================================================================
// MDTA: attn[b,h] = softmax(temp[h] * qn @ kn^T)  via WMMA K-reduction
// QK points at the 192-channel qkv buffer (q=+0, k=+64ch)
// =====================================================================
__global__ __launch_bounds__(256) void k_attn_qk(
    const _Float16* __restrict__ QK, const float* __restrict__ qinv,
    const float* __restrict__ kinv, const float* __restrict__ temp,
    float* __restrict__ attn) {
  int bh = blockIdx.x;
  int b = bh >> 2, h = bh & 3;
  const _Float16* q = QK + ((size_t)b * 192 + h * 16) * HWN;
  const _Float16* k = QK + ((size_t)b * 192 + 64 + h * 16) * HWN;
  const float* qi = qinv + b * 64 + h * 16;
  const float* ki = kinv + b * 64 + h * 16;
  int wave = threadIdx.x >> 5, lane = threadIdx.x & 31;
  int row = lane & 15;
  float qs = qi[row], ks = ki[row];
  v8f acc = {0.f, 0.f, 0.f, 0.f, 0.f, 0.f, 0.f, 0.f};
  for (int it = 0; it < 64; ++it) {
    int n0 = (it * 8 + wave) * 32;
    v16h af, bf;
#pragma unroll
    for (int hh = 0; hh < 16; ++hh) {
      int Ka = (lane < 16) ? ((hh < 8) ? hh : hh + 8) : ((hh < 8) ? hh + 8 : hh + 16);
      af[hh] = (_Float16)((float)q[(size_t)row * HWN + n0 + Ka] * qs);
      int Kb = (lane < 16) ? ((hh < 8) ? hh : hh + 8) : ((hh < 8) ? hh + 8 : hh + 16);
      bf[hh] = (_Float16)((float)k[(size_t)row * HWN + n0 + Kb] * ks);
    }
    acc = __builtin_amdgcn_wmma_f32_16x16x32_f16(false, af, false, bf, (short)0,
                                                 acc, false, false);
  }
  __shared__ float part[8 * 256];
  int moff = (lane >= 16) ? 8 : 0;
#pragma unroll
  for (int r = 0; r < 8; ++r)
    part[wave * 256 + (r + moff) * 16 + (lane & 15)] = acc[r];
  __syncthreads();
  float sum = 0.f;
  for (int w2 = 0; w2 < 8; ++w2) sum += part[w2 * 256 + threadIdx.x];
  sum *= temp[h];
  __syncthreads();
  part[threadIdx.x] = sum;
  __syncthreads();
  if (threadIdx.x < 16) {
    int r = threadIdx.x;
    float mx = -1e30f;
    for (int j = 0; j < 16; ++j) mx = fmaxf(mx, part[r * 16 + j]);
    float e[16], se = 0.f;
    for (int j = 0; j < 16; ++j) { e[j] = expf(part[r * 16 + j] - mx); se += e[j]; }
    for (int j = 0; j < 16; ++j) attn[bh * 256 + r * 16 + j] = e[j] / se;
  }
}

// out[b, h*16+c, n] = sum_d attn[b,h,c,d] * v[b, h*16+d, n]
__global__ __launch_bounds__(256) void k_attn_av(
    const _Float16* __restrict__ QK, const float* __restrict__ attn,
    _Float16* __restrict__ Y) {
  int bh = blockIdx.y;
  int b = bh >> 2, h = bh & 3;
  __shared__ float a[256];
  a[threadIdx.x] = attn[bh * 256 + threadIdx.x];
  __syncthreads();
  int p = blockIdx.x * 256 + threadIdx.x;
  const _Float16* v = QK + ((size_t)b * 192 + 128 + h * 16) * HWN + p;
  float vv[16];
#pragma unroll
  for (int d = 0; d < 16; ++d) vv[d] = (float)v[(size_t)d * HWN];
#pragma unroll
  for (int c = 0; c < 16; ++c) {
    float s = 0.f;
#pragma unroll
    for (int d = 0; d < 16; ++d) s += a[c * 16 + d] * vv[d];
    Y[((size_t)b * 64 + h * 16 + c) * HWN + p] = (_Float16)s;
  }
}

// gelu(x1)*x2 over a 256ch buffer -> 128ch buffer
__global__ __launch_bounds__(256) void k_gelumul(const _Float16* __restrict__ X,
                                                 _Float16* __restrict__ Y) {
  int bc = blockIdx.y;
  int b = bc / 128, c = bc % 128;
  int p = blockIdx.x * 256 + threadIdx.x;
  float x1 = (float)X[((size_t)b * 256 + c) * HWN + p];
  float x2 = (float)X[((size_t)b * 256 + 128 + c) * HWN + p];
  float g = 0.5f * x1 * (1.f + erff(x1 * 0.70710678f));
  Y[((size_t)b * 128 + c) * HWN + p] = (_Float16)(g * x2);
}

// PDMB dynamic kernel generation: hk=lrelu(dvec@k1^T); kern=hk@k2^T
__global__ __launch_bounds__(256) void k_pdmbgen(const float* __restrict__ dvec,
                                                 const float* __restrict__ k1,
                                                 const float* __restrict__ k2,
                                                 float* __restrict__ kern) {
  __shared__ float hk[BATCH * 32];
  int t = threadIdx.x;
  {
    int b = t >> 5, j = t & 31;
    float s = 0.f;
    for (int i = 0; i < 32; ++i) s += dvec[b * 32 + i] * k1[j * 32 + i];
    hk[b * 32 + j] = (s >= 0.f) ? s : 0.1f * s;
  }
  __syncthreads();
  for (int o = t; o < BATCH * 288; o += 256) {
    int b = o / 288, j = o % 288;
    float s = 0.f;
    for (int i = 0; i < 32; ++i) s += hk[b * 32 + i] * k2[j * 32 + i];
    kern[o] = s;
  }
}

// final: out = y + x_input  (f32)
__global__ __launch_bounds__(256) void k_final(const _Float16* __restrict__ y,
                                               const float* __restrict__ x,
                                               float* __restrict__ out, int n) {
  int i = blockIdx.x * 256 + threadIdx.x;
  if (i < n) out[i] = (float)y[i] + x[i];
}

// =====================================================================
// host orchestration
// =====================================================================
extern "C" void kernel_launch(void* const* d_in, const int* in_sizes, int n_in,
                              void* d_out, int out_size, void* d_ws, size_t ws_size,
                              hipStream_t stream) {
  (void)in_sizes; (void)n_in; (void)out_size; (void)ws_size;
  auto FP = [&](int i) { return reinterpret_cast<const float*>(d_in[i]); };

  // ---- flattened param indices (jax pytree order: sorted dict keys) ----
  // 0:x 1:weights | attn: 2 a0w 3 a0b 4 a1w 5 a1b 6 ffn_dw 7 ffn_in 8 ffn_out
  // 9 n1 10 n2 11 proj 12 qkv 13 qkv_dw 14 temp 15 dww 16 dwb 17 scaw 18 scab
  // blocks[i] base = 19 + 32*i (layout in comments below)
  const int A0W=2, A0B=3, A1W=4, A1B=5, FFNDW=6, FFNIN=7, FFNOUT=8,
            N1=9, N2=10, PROJ=11, QKVW=12, QKVDW=13, TEMP=14,
            ADWW=15, ADWB=16, ASCAW=17, ASCAB=18;

  char* cur = (char*)d_ws;
  auto alloc = [&](size_t bytes) -> char* {
    char* p = cur;
    cur += (bytes + 255) & ~(size_t)255;
    return p;
  };
  auto hbuf = [&](int C) { return (_Float16*)alloc((size_t)BATCH * C * HWN * 2); };

  _Float16* X0  = hbuf(64);     // input in f16
  _Float16* CAT = hbuf(256);    // dist concat (block outputs live here)
  _Float16* T0  = hbuf(64);
  _Float16* T1  = hbuf(64);
  _Float16* T2  = hbuf(64);
  _Float16* T3  = hbuf(64);
  _Float16* QKV = hbuf(192);    // also reused as 128ch gelu buffer
  _Float16* FFN = hbuf(256);    // also reused as 192ch qkv-dw buffer
  float* pooled  = (float*)alloc(BATCH * 64 * 4);
  float* svec    = (float*)alloc(BATCH * 64 * 4);
  float* qinv    = (float*)alloc(BATCH * 64 * 4);
  float* kinv    = (float*)alloc(BATCH * 64 * 4);
  float* attnB   = (float*)alloc(BATCH * HEADS * 256 * 4);
  float* kern    = (float*)alloc(BATCH * 288 * 4);

  auto conv1x1 = [&](const _Float16* Xv, int XCb, int Cin, const float* W,
                     const float* Bb, int Cout, _Float16* Yv, int YCb, int act) {
    int Kt = (Cin + 31) / 32;
    int MT = Cout / 16;
    size_t tot = (size_t)MT * Kt * 512;
    _Float16* ap = (_Float16*)alloc(tot * 2);
    k_pack<<<dim3((unsigned)((tot + 255) / 256)), 256, 0, stream>>>(W, ap, Cout, Cin, Kt);
    dim3 g(HWN / 128, BATCH);
    switch (MT) {
      case 1:  k_conv1x1<1><<<g, 256, 0, stream>>>(Xv, XCb, ap, Kt, Cin, Bb, Yv, YCb, act); break;
      case 2:  k_conv1x1<2><<<g, 256, 0, stream>>>(Xv, XCb, ap, Kt, Cin, Bb, Yv, YCb, act); break;
      case 4:  k_conv1x1<4><<<g, 256, 0, stream>>>(Xv, XCb, ap, Kt, Cin, Bb, Yv, YCb, act); break;
      case 12: k_conv1x1<12><<<g, 256, 0, stream>>>(Xv, XCb, ap, Kt, Cin, Bb, Yv, YCb, act); break;
      default: k_conv1x1<16><<<g, 256, 0, stream>>>(Xv, XCb, ap, Kt, Cin, Bb, Yv, YCb, act); break;
    }
  };
  auto dw3x3 = [&](const _Float16* Xv, int XCb, const float* W, const float* Bb,
                   _Float16* Yv, int YCb, int C, int act, int dyn) {
    k_dw3x3<<<dim3(HWN / 256, BATCH * C), 256, 0, stream>>>(Xv, XCb, W, Bb, Yv, YCb, C, act, dyn);
  };
  auto axpy = [&](_Float16* D, int DCb, const _Float16* S, int SCb, int C, int add) {
    k_axpy<<<dim3(HWN / 256, BATCH * C), 256, 0, stream>>>(D, DCb, S, SCb, C, add);
  };

  // ---- input -> f16 ----
  {
    int n = BATCH * 64 * HWN;
    k_cvt<<<dim3((n + 255) / 256), 256, 0, stream>>>(FP(0), X0, n);
  }

  // ================= 4 LDMB blocks =================
  for (int i = 0; i < 4; ++i) {
    int bb = 19 + 32 * i;
    const float *c1w0 = FP(bb+0), *c1b0 = FP(bb+1), *c1dw = FP(bb+2), *c1db = FP(bb+3),
                *c1w2 = FP(bb+4), *c1b2 = FP(bb+5),
                *c2w0 = FP(bb+6), *c2b0 = FP(bb+7), *c2dw = FP(bb+8), *c2db = FP(bb+9),
                *c2w2 = FP(bb+10), *c2b2 = FP(bb+11),
                *hc0w = FP(bb+12), *hc0b = FP(bb+13), *hc1w = FP(bb+14), *hc1b = FP(bb+15),
                *hd0w = FP(bb+16), *hd0b = FP(bb+17), *hd1w = FP(bb+18), *hd1b = FP(bb+19),
                *hpcw = FP(bb+20), *hpcb = FP(bb+21), *hk1 = FP(bb+22), *hk2 = FP(bb+23),
                *lc0w = FP(bb+24), *lc0b = FP(bb+25), *lc3w = FP(bb+26), *lc3b = FP(bb+27),
                *ldww = FP(bb+28), *ldwb = FP(bb+29), *lscw = FP(bb+30), *lscb = FP(bb+31);

    const _Float16* xin = (i == 0) ? X0 : (CAT + (size_t)(i - 1) * 64 * HWN);
    int xinCb = (i == 0) ? 64 : 256;

    // conv_seq conv1: T0 = lrelu(1x1(lrelu(1x1(x)) -> dw))
    conv1x1(xin, xinCb, 64, c1w0, c1b0, 64, T1, 64, 1);
    dw3x3(T1, 64, c1dw, c1db, T2, 64, 64, 0, 0);
    conv1x1(T2, 64, 64, c1w2, c1b2, 64, T0, 64, 1);

    // ---- branch1 on T0[0:32] ----
    conv1x1(T0, 64, 32, hc0w, hc0b, 32, T1, 64, 1);
    dw3x3(T1, 64, hd0w, hd0b, T2, 64, 32, 1, 0);
    k_pdmbgen<<<1, 256, 0, stream>>>(FP(1), hk1, hk2, kern);
    dw3x3(T2, 64, kern, nullptr, T1, 64, 32, 1, 1);       // dyn dw + lrelu
    conv1x1(T1, 64, 32, hpcw, hpcb, 32, T2, 64, 0);
    conv1x1(T2, 64, 32, hc1w, hc1b, 32, T1, 64, 1);
    dw3x3(T1, 64, hd1w, hd1b, T3, 64, 32, 1, 0);          // branch1 -> T3[0:32]

    // ---- branch2 on T0[32:64] (f1=ch32..47, f2=ch48..63) ----
    conv1x1(T0 + (size_t)32 * HWN, 64, 16, lc0w, lc0b, 16, T1 + (size_t)32 * HWN, 64, 1);
    dw3x3(T1 + (size_t)32 * HWN, 64, ldww, ldwb, T2, 64, 16, 1, 0);
    k_conv3x3<<<dim3(HWN / 256, BATCH * 16), 256, 0, stream>>>(
        T2, 64, lc3w, lc3b, T2 + (size_t)16 * HWN, 64, 16, 16, 1);
    k_gap<<<dim3(BATCH * 16), 256, 0, stream>>>(T2 + (size_t)16 * HWN, 64, 16, pooled, 0);
    k_scavec<<<dim3(1), 256, 0, stream>>>(pooled, lscw, lscb, svec, 16, 16);
    k_scale<<<dim3(HWN / 256, BATCH * 16), 256, 0, stream>>>(T2 + (size_t)16 * HWN, 64, 16, svec, 0);
    axpy(T3 + (size_t)32 * HWN, 64, T2 + (size_t)16 * HWN, 64, 16, 0); // sca(y)
    axpy(T3 + (size_t)48 * HWN, 64, T0 + (size_t)48 * HWN, 64, 16, 0); // f2

    // conv_seq conv2 -> CAT slot i ; residual += xin
    _Float16* slot = CAT + (size_t)i * 64 * HWN;
    conv1x1(T3, 64, 64, c2w0, c2b0, 64, T1, 64, 1);
    dw3x3(T1, 64, c2dw, c2db, T2, 64, 64, 0, 0);
    conv1x1(T2, 64, 64, c2w2, c2b2, 64, slot, 256, 1);
    axpy(slot, 256, xin, xinCb, 64, 1);
  }

  // ================= attention tail =================
  conv1x1(CAT, 256, 256, FP(A0W), FP(A0B), 64, T0, 64, 1);
  conv1x1(T0, 64, 64, FP(A1W), FP(A1B), 64, T1, 64, 1);
  dw3x3(T1, 64, FP(ADWW), FP(ADWB), T0, 64, 64, 1, 0);
  k_gap<<<dim3(BATCH * 64), 256, 0, stream>>>(T0, 64, 64, pooled, 0);
  k_scavec<<<dim3((BATCH * 64 + 255) / 256), 256, 0, stream>>>(pooled, FP(ASCAW), FP(ASCAB), svec, 64, 64);
  k_scale<<<dim3(HWN / 256, BATCH * 64), 256, 0, stream>>>(T0, 64, 64, svec, 1); // lrelu(sca(x))

  // ---- CSA: MDTA ----
  k_ln<<<dim3(HWN / 256, BATCH), 256, 0, stream>>>(T0, FP(N1), T1, 64);
  conv1x1(T1, 64, 64, FP(QKVW), nullptr, 192, QKV, 192, 0);
  _Float16* QKN = FFN;  // reuse 256ch buffer as the 192ch dw output
  dw3x3(QKV, 192, FP(QKVDW), nullptr, QKN, 192, 192, 0, 0);
  k_gap<<<dim3(BATCH * 64), 256, 0, stream>>>(QKN, 192, 64, qinv, 1);
  k_gap<<<dim3(BATCH * 64), 256, 0, stream>>>(QKN + (size_t)64 * HWN, 192, 64, kinv, 1);
  k_attn_qk<<<dim3(BATCH * HEADS), 256, 0, stream>>>(QKN, qinv, kinv, FP(TEMP), attnB);
  k_attn_av<<<dim3(HWN / 256, BATCH * HEADS), 256, 0, stream>>>(QKN, attnB, T2);
  conv1x1(T2, 64, 64, FP(PROJ), nullptr, 64, T1, 64, 0);
  axpy(T0, 64, T1, 64, 64, 1);   // y = y + mdta

  // ---- CSA: GDFN ----
  k_ln<<<dim3(HWN / 256, BATCH), 256, 0, stream>>>(T0, FP(N2), T1, 64);
  conv1x1(T1, 64, 64, FP(FFNIN), nullptr, 256, FFN, 256, 0);
  dw3x3(FFN, 256, FP(FFNDW), nullptr, CAT, 256, 256, 0, 0);  // CAT is free now
  _Float16* G = QKV;             // reuse as 128ch buffer
  k_gelumul<<<dim3(HWN / 256, BATCH * 128), 256, 0, stream>>>(CAT, G);
  conv1x1(G, 128, 128, FP(FFNOUT), nullptr, 64, T1, 64, 0);
  axpy(T0, 64, T1, 64, 64, 1);   // y = y + ffn

  // ---- final residual with original input (f32 out) ----
  {
    int n = BATCH * 64 * HWN;
    k_final<<<dim3((n + 255) / 256), 256, 0, stream>>>(T0, FP(0), (float*)d_out, n);
  }
}
